// AutoCorrelation_7705171329202
// MI455X (gfx1250) — matrix operand order repaired and checked by reading, not compiled, not run on hardware
//
#include <hip/hip_runtime.h>
#include <hip/hip_bf16.h>

typedef __attribute__((ext_vector_type(16))) __bf16 v16bf;
typedef __attribute__((ext_vector_type(8)))  __bf16 bf16x8;
typedef __attribute__((ext_vector_type(2)))  __bf16 bf16x2;
typedef __attribute__((ext_vector_type(8)))  float  v8f;

// ---- fp32 -> bf16 (lowers to v_cvt_pk_bf16_f32, asm-verified) -------------
__device__ __forceinline__ bf16x2 cvtpk2(float lo, float hi) {
#if __has_builtin(__builtin_amdgcn_cvt_pk_bf16_f32)
  return __builtin_amdgcn_cvt_pk_bf16_f32(lo, hi);
#else
  bf16x2 r; r[0] = (__bf16)lo; r[1] = (__bf16)hi;
  return r;
#endif
}
__device__ __forceinline__ __bf16 f2bf(float f) { return (__bf16)f; }

// native v_exp_f32: args here are <= 0; sub-(-126) underflow to ~0 is exactly
// what softmax wants, so skip libm's denormal-range fixup scaffolding.
__device__ __forceinline__ float fexp2(float x) {
#if __has_builtin(__builtin_amdgcn_exp2f)
  return __builtin_amdgcn_exp2f(x);
#else
  return exp2f(x);
#endif
}

__device__ __forceinline__ v16bf cat8(bf16x8 lo, bf16x8 hi) {
  return __builtin_shufflevector(lo, hi, 0,1,2,3,4,5,6,7,8,9,10,11,12,13,14,15);
}

// max-reduce across each 16-lane half (xor swizzle ladder)
#if __has_builtin(__builtin_amdgcn_ds_swizzle)
#define SWZ_XMAX(v, imm) \
  fmaxf((v), __builtin_bit_cast(float, __builtin_amdgcn_ds_swizzle(__builtin_bit_cast(int, (v)), (imm))))
__device__ __forceinline__ float rowmax16(float v) {
  v = SWZ_XMAX(v, 0x041F);
  v = SWZ_XMAX(v, 0x081F);
  v = SWZ_XMAX(v, 0x101F);
  v = SWZ_XMAX(v, 0x201F);
  return v;
}
#else
__device__ __forceinline__ float rowmax16(float v) {
#pragma unroll
  for (int m = 1; m < 16; m <<= 1) v = fmaxf(v, __shfl_xor(v, m, 32));
  return v;
}
#endif

// ---- prep kernel 1: K [B,S,H,64] f32 -> [B,H,S,64] bf16 -------------------
__global__ __launch_bounds__(256)
void pack_k_kernel(const float* __restrict__ src, __bf16* __restrict__ dst)
{
  const int idx = blockIdx.x * 256 + threadIdx.x;   // pair index
  const int e2 = idx & 31;            // e = 2*e2
  int t = idx >> 5;                   // row over [B,X,H], h innermost
  const int h  = t & 15;  t >>= 4;
  const int x  = t & 2047; t >>= 11;
  const int b  = t;
  const float2 v = *(const float2*)(src + ((((size_t)b*2048 + x)*16 + h)*64) + 2*e2);
  *(bf16x2*)(dst + ((((size_t)b*16 + h)*2048 + x)*64) + 2*e2) = cvtpk2(v.x, v.y);
}

// ---- prep kernel 2: V [B,S,H,64] f32 -> V^T [B,H,64,S] bf16 ---------------
__global__ __launch_bounds__(256)
void transpose_v_kernel(const float* __restrict__ V, __bf16* __restrict__ Vt)
{
  __shared__ __bf16 T[32][33];
  const int b  = blockIdx.z >> 4;
  const int h  = blockIdx.z & 15;
  const int s0 = blockIdx.x * 32;
  const int d0 = blockIdx.y * 32;
  const int tx = threadIdx.x;          // 0..31
  const int ty = threadIdx.y;          // 0..7
#pragma unroll
  for (int i = 0; i < 4; ++i) {
    const int r = ty + 8 * i;          // s offset in tile
    T[r][tx] = f2bf(V[(((size_t)b*2048 + (s0+r))*16 + h)*64 + d0 + tx]);
  }
  __syncthreads();
#pragma unroll
  for (int i = 0; i < 4; ++i) {
    const int r = ty + 8 * i;          // d offset in tile
    Vt[(((size_t)b*16 + h)*64 + (d0+r))*2048 + s0 + tx] = T[tx][r];
  }
}

// ---- main kernel: flash-attention fwd over bf16 K / V^T, 64-key tiles -----
// Q:[B,L,H,E] f32, Kb:[B,H,S,E] bf16, Vtb:[B,H,D,S] bf16 -> O:[B,L,H,D] f32
__global__ __launch_bounds__(128)
void fa_fwd_kernel(const float* __restrict__ Qg, const __bf16* __restrict__ Kb,
                   const __bf16* __restrict__ Vtb, float* __restrict__ Og)
{
  constexpr int L = 2048, S = 2048, H = 16, E = 64, D = 64;
  constexpr float LOG2E = 1.4426950408889634f;

  const int tid  = threadIdx.x;
  const int wave = tid >> 5;
  const int lane = tid & 31;
  const int lh   = lane & 15;
  const int lhi  = lane >> 4;

  const int b = blockIdx.y >> 4;
  const int h = blockIdx.y & 15;
  const int qrow0 = blockIdx.x * 64 + wave * 16;

  // per-wave P staging (16x64 bf16 = 2KB per wave); no block barriers
  __shared__ __align__(16) __bf16 Pl[4][16][64];

  // ---- Q A-operands (fp32 source, scale 1/8 folded in) ----
  v16bf qa[2];
  {
    const float* qrow = Qg + (((size_t)b * L + (qrow0 + lh)) * H + h) * E;
#pragma unroll
    for (int c = 0; c < 2; ++c) {
      const int e0 = 32 * c + 8 * lhi;
      float4 f0 = *(const float4*)(qrow + e0);
      float4 f1 = *(const float4*)(qrow + e0 + 4);
      float4 f2 = *(const float4*)(qrow + e0 + 16);
      float4 f3 = *(const float4*)(qrow + e0 + 20);
      union { v16bf v; bf16x2 p[8]; } u;
      u.p[0] = cvtpk2(f0.x*0.125f, f0.y*0.125f);
      u.p[1] = cvtpk2(f0.z*0.125f, f0.w*0.125f);
      u.p[2] = cvtpk2(f1.x*0.125f, f1.y*0.125f);
      u.p[3] = cvtpk2(f1.z*0.125f, f1.w*0.125f);
      u.p[4] = cvtpk2(f2.x*0.125f, f2.y*0.125f);
      u.p[5] = cvtpk2(f2.z*0.125f, f2.w*0.125f);
      u.p[6] = cvtpk2(f3.x*0.125f, f3.y*0.125f);
      u.p[7] = cvtpk2(f3.z*0.125f, f3.w*0.125f);
      qa[c] = u.v;
    }
  }

  // all-ones B operand: rowsum(P) via WMMA; pin in VGPRs (avoid per-iter remat)
  v16bf onesb;
#pragma unroll
  for (int i = 0; i < 16; ++i)
    onesb[i] = __builtin_bit_cast(__bf16, (unsigned short)0x3F80);
  asm("" : "+v"(onesb));

  v8f z8;
#pragma unroll
  for (int j = 0; j < 8; ++j) z8[j] = 0.f;

  v8f oo[4];
  float m[8], l[8];
#pragma unroll
  for (int j = 0; j < 8; ++j) {
    m[j] = -__builtin_inff(); l[j] = 0.f;
    oo[0][j] = 0.f; oo[1][j] = 0.f; oo[2][j] = 0.f; oo[3][j] = 0.f;
  }

  const __bf16* Kbh = Kb  + ((size_t)blockIdx.y) * S * E;  // [s][e]
  const __bf16* Vbh = Vtb + ((size_t)blockIdx.y) * D * S;  // [d][s]

  for (int s0 = 0; s0 < S; s0 += 64) {
    // ---- K operands: 4 key-subtiles x 2 e-chunks (16x b128) ----
    v16bf kb[4][2];
#pragma unroll
    for (int t = 0; t < 4; ++t)
#pragma unroll
      for (int c = 0; c < 2; ++c) {
        const bf16x8* kp = (const bf16x8*)(Kbh + (unsigned)(s0 + 16*t + lh) * E
                                               + 32*c + 8*lhi);
        kb[t][c] = cat8(kp[0], kp[2]);
      }

    // ---- scores: 4 x (2-chunk WMMA chain) ----
    v8f sacc[4];
#pragma unroll
    for (int t = 0; t < 4; ++t) {
      sacc[t] = __builtin_amdgcn_wmma_f32_16x16x32_bf16(
          false, qa[0], false, kb[t][0], (short)0, z8, false, false);
      sacc[t] = __builtin_amdgcn_wmma_f32_16x16x32_bf16(
          false, qa[1], false, kb[t][1], (short)0, sacc[t], false, false);
    }

    // ---- V operands issued now; latency hides behind softmax ----
    v16bf vb[4][2];
#pragma unroll
    for (int t = 0; t < 4; ++t)
#pragma unroll
      for (int u = 0; u < 2; ++u) {
        const bf16x8* vp = (const bf16x8*)(Vbh + (unsigned)(16*t + lh) * S
                                               + (unsigned)s0 + 32*u + 8*lhi);
        vb[t][u] = cat8(vp[0], vp[2]);
      }

    if (s0 + 64 < S) {
      __builtin_prefetch((const void*)(Kbh + (size_t)(s0 + 64 + lh) * E), 0, 1);
      __builtin_prefetch((const void*)(Vbh + (size_t)lane * S + s0 + 64), 0, 1);
    }

    // ---- one softmax pass over all 64 columns ----
    float p[4][8], cr[8];
#pragma unroll
    for (int j = 0; j < 8; ++j) {
      float cand = fmaxf(fmaxf(sacc[0][j], sacc[1][j]),
                         fmaxf(sacc[2][j], sacc[3][j]));
      float mn = fmaxf(m[j], rowmax16(cand));
      const float mnl = mn * LOG2E;
      p[0][j] = fexp2(sacc[0][j] * LOG2E - mnl);
      p[1][j] = fexp2(sacc[1][j] * LOG2E - mnl);
      p[2][j] = fexp2(sacc[2][j] * LOG2E - mnl);
      p[3][j] = fexp2(sacc[3][j] * LOG2E - mnl);
      cr[j] = fexp2(m[j] * LOG2E - mnl);
      m[j]  = mn;
      oo[0][j] *= cr[j]; oo[1][j] *= cr[j]; oo[2][j] *= cr[j]; oo[3][j] *= cr[j];
    }

    // ---- P (16x64, C layout) -> LDS -> two A-layout chunks ----
#pragma unroll
    for (int t = 0; t < 4; ++t)
#pragma unroll
      for (int j = 0; j < 8; ++j)
        Pl[wave][8 * lhi + j][16 * t + lh] = f2bf(p[t][j]);
    asm volatile("s_wait_dscnt 0x0" ::: "memory");
    v16bf pa[2];
#pragma unroll
    for (int u = 0; u < 2; ++u) {
      bf16x8 alo = *(const bf16x8*)&Pl[wave][lh][32*u + 8 * lhi];
      bf16x8 ahi = *(const bf16x8*)&Pl[wave][lh][32*u + 16 + 8 * lhi];
      pa[u] = cat8(alo, ahi);
    }

    // ---- row sums of P via ones-WMMA chain ----
    v8f ls = __builtin_amdgcn_wmma_f32_16x16x32_bf16(
        false, pa[0], false, onesb, (short)0, z8, false, false);
    ls = __builtin_amdgcn_wmma_f32_16x16x32_bf16(
        false, pa[1], false, onesb, (short)0, ls, false, false);
#pragma unroll
    for (int j = 0; j < 8; ++j) l[j] = l[j] * cr[j] + ls[j];

    // ---- O += P * V : 4 d-tiles x 2-chunk chains ----
#pragma unroll
    for (int t = 0; t < 4; ++t) {
      oo[t] = __builtin_amdgcn_wmma_f32_16x16x32_bf16(
          false, pa[0], false, vb[t][0], (short)0, oo[t], false, false);
      oo[t] = __builtin_amdgcn_wmma_f32_16x16x32_bf16(
          false, pa[1], false, vb[t][1], (short)0, oo[t], false, false);
    }
  }

  // ---- epilogue ----
#pragma unroll
  for (int j = 0; j < 8; ++j) {
    const float inv = 1.0f / l[j];
    const int row = qrow0 + 8 * lhi + j;
    float* orow = Og + (((size_t)b * L + row) * H + h) * D;
#pragma unroll
    for (int t = 0; t < 4; ++t) orow[16 * t + lh] = oo[t][j] * inv;
  }
}

extern "C" void kernel_launch(void* const* d_in, const int* in_sizes, int n_in,
                              void* d_out, int out_size, void* d_ws, size_t ws_size,
                              hipStream_t stream) {
  (void)in_sizes; (void)n_in; (void)out_size; (void)ws_size;
  const float* Q = (const float*)d_in[0];
  const float* K = (const float*)d_in[1];
  const float* V = (const float*)d_in[2];
  float* O = (float*)d_out;
  constexpr int B = 4, L = 2048, S = 2048, H = 16, E = 64, D = 64;

  __bf16* Kbf = (__bf16*)d_ws;                        // [B,H,S,E] bf16, 16 MiB
  __bf16* Vtb = Kbf + (size_t)B * H * S * E;          // [B,H,D,S] bf16, 16 MiB

  pack_k_kernel<<<(B * S * H * E / 2) / 256, 256, 0, stream>>>(K, Kbf);
  transpose_v_kernel<<<dim3(S / 32, D / 32, B * H), dim3(32, 8), 0, stream>>>(V, Vtb);

  fa_fwd_kernel<<<dim3(L / 64, B * H), 128, 0, stream>>>(Q, Kbf, Vtb, O);
}
